// attention_block_90228672955012
// MI455X (gfx1250) — compile-verified, matrix operand order checked by reading, not run on hardware
//
#include <hip/hip_runtime.h>
#include <hip/hip_bf16.h>

// ---------------------------------------------------------------------------
// Fused attention block for MI455X (gfx1250, wave32, WMMA f16 path).
//   x:(4,2048,768) f32 -> out:(4,2048,768) f32
// cast->f16 | QKV GEMM (LDS-staged B, 32x64/wave) | flash attention
// (TDM tensor_load_to_lds K/V staging, 32 queries/wave, online softmax) |
// proj GEMM.  Scores never touch HBM.
// ---------------------------------------------------------------------------

typedef __attribute__((ext_vector_type(16))) _Float16 v16h;
typedef __attribute__((ext_vector_type(8)))  _Float16 v8h;
typedef __attribute__((ext_vector_type(4)))  _Float16 v4h;
typedef __attribute__((ext_vector_type(8)))  float    v8f;
typedef __attribute__((ext_vector_type(4)))  unsigned int u32x4;
typedef __attribute__((ext_vector_type(8)))  int i32x8;
typedef __attribute__((ext_vector_type(4)))  int i32x4;

#define HIDDEN   768
#define NHEADS   12
#define HDIM     64
#define SEQ      2048
#define BATCH    4
#define MTOT     (BATCH * SEQ)      // 8192
#define QKV_OUT  (3 * HIDDEN)       // 2304

#if __has_builtin(__builtin_amdgcn_tensor_load_to_lds)
#define HAVE_TDM 1
#endif

static __device__ __forceinline__ v8f wmma_f16(v16h a, v16h b, v8f c) {
    return __builtin_amdgcn_wmma_f32_16x16x32_f16(false, a, false, b,
                                                  (short)0, c, false, false);
}

// One 16x32 f16 fragment slice for this lane (A or B operand).
// khalf = (lane>=16)?8:0; elements 0..7 <- K khalf.., 8..15 <- K 16+khalf..
static __device__ __forceinline__ v16h load_frag(const _Float16* base, int khalf) {
    v8h lo = *(const v8h*)(base + khalf);
    v8h hi = *(const v8h*)(base + 16 + khalf);
    v16h r;
#pragma unroll
    for (int i = 0; i < 8; ++i) { r[i] = lo[i]; r[i + 8] = hi[i]; }
    return r;
}

#ifdef HAVE_TDM
// Issue one 2D TDM tile load into LDS. g1 carries data_size/pad/tile/stride;
// group0 carries count=1, LDS addr, 57-bit global addr, type=2 (image).
static __device__ __forceinline__ void tdm_load_2d(unsigned lds_addr,
                                                   const void* gptr,
                                                   i32x8 g1) {
    unsigned long long ga = (unsigned long long)(size_t)gptr;
    u32x4 g0;
    g0[0] = 1u;                               // count=1, user mode
    g0[1] = lds_addr;                         // LDS byte address
    g0[2] = (unsigned)ga;                     // global_addr[31:0]
    g0[3] = ((unsigned)(ga >> 32) & 0x01FFFFFFu) | 0x80000000u; // [56:32]|type=2
    i32x4 gz = {};
#if __clang_major__ >= 23
    i32x8 gz8 = {};
    __builtin_amdgcn_tensor_load_to_lds(g0, g1, gz, gz, gz8, 0);
#else
    __builtin_amdgcn_tensor_load_to_lds(g0, g1, gz, gz, 0);
#endif
}
#endif

// ---------------------------------------------------------------------------
// Kernel 1: f32 -> f16 cast
// ---------------------------------------------------------------------------
__global__ void cast_f32_to_f16(const float* __restrict__ s,
                                _Float16* __restrict__ d, int n) {
    int i = (blockIdx.x * blockDim.x + threadIdx.x) * 4;
    if (i < n) {
        float4 v = *(const float4*)(s + i);
        v4h r;
        r[0] = (_Float16)v.x; r[1] = (_Float16)v.y;
        r[2] = (_Float16)v.z; r[3] = (_Float16)v.w;
        *(v4h*)(d + i) = r;
    }
}

// ---------------------------------------------------------------------------
// Kernel 2: QKV GEMM. M=8192, N=2304, K=768.
// Block: 8 waves, 256 rows x 64 cols; B tile staged in LDS (double buffer);
// each wave computes 32x64 (8 WMMAs per 32-K step).
// ---------------------------------------------------------------------------
__global__ void __launch_bounds__(256, 1)
qkv_gemm(const _Float16* __restrict__ xh,
         const _Float16* __restrict__ wh,
         const float*    __restrict__ bias,
         _Float16* __restrict__ qh,
         _Float16* __restrict__ kh,
         _Float16* __restrict__ vth) {
    __shared__ _Float16 bsh[2][64][40];         // [buf][col][k] pitch 80B
    const int tid   = threadIdx.x;
    const int lane  = tid & 31;
    const int wave  = tid >> 5;
    const int brow  = blockIdx.x % (MTOT / 256);        // 32
    const int bcol  = blockIdx.x / (MTOT / 256);        // 36
    const int m0    = brow * 256 + wave * 32;
    const int o0    = bcol * 64;
    const int lmod  = lane & 15;
    const int khalf = (lane >> 4) * 8;

    const int scol = tid >> 2;
    const int skc  = (tid & 3) * 8;
    const _Float16* gB = wh + (size_t)(o0 + scol) * HIDDEN + skc;
    v8h breg = *(const v8h*)gB;

    const _Float16* ar0 = xh + (size_t)(m0 + lmod) * HIDDEN;
    const _Float16* ar1 = ar0 + (size_t)16 * HIDDEN;

    v8f acc[2][4] = {};
    for (int kt = 0; kt < 24; ++kt) {
        const int buf = kt & 1;
        *(v8h*)&bsh[buf][scol][skc] = breg;      // publish staged tile
        if (kt < 23) breg = *(const v8h*)(gB + (kt + 1) * 32);  // load ahead
        __syncthreads();
        const int k = kt * 32;
        v16h a0 = load_frag(ar0 + k, khalf);
        v16h a1 = load_frag(ar1 + k, khalf);
#pragma unroll
        for (int t = 0; t < 4; ++t) {
            v16h b = load_frag(&bsh[buf][t * 16 + lmod][0], khalf);
            acc[0][t] = wmma_f16(a0, b, acc[0][t]);
            acc[1][t] = wmma_f16(a1, b, acc[1][t]);
        }
    }

    const int b  = m0 / SEQ;
    const int n0 = m0 % SEQ;
#pragma unroll
    for (int mt = 0; mt < 2; ++mt) {
        const int rbase = n0 + mt * 16 + ((lane >> 4) * 8);
#pragma unroll
        for (int t = 0; t < 4; ++t) {
            const int o   = o0 + t * 16 + lmod;
            const int seg = o / HIDDEN;          // uniform per (t)
            const int oo  = o % HIDDEN;
            const int h   = oo / HDIM;
            const int d   = oo % HDIM;
            const int bh  = b * NHEADS + h;
            const float bv = bias[o];
            if (seg == 0) {
                _Float16* dst = qh + ((size_t)(bh * SEQ + rbase)) * HDIM + d;
#pragma unroll
                for (int c = 0; c < 8; ++c)
                    dst[c * HDIM] = (_Float16)((acc[mt][t][c] + bv) * 0.125f);
            } else if (seg == 1) {
                _Float16* dst = kh + ((size_t)(bh * SEQ + rbase)) * HDIM + d;
#pragma unroll
                for (int c = 0; c < 8; ++c)
                    dst[c * HDIM] = (_Float16)(acc[mt][t][c] + bv);
            } else {
                v8h pk;
#pragma unroll
                for (int c = 0; c < 8; ++c)
                    pk[c] = (_Float16)(acc[mt][t][c] + bv);
                *(v8h*)(vth + ((size_t)(bh * HDIM + d)) * SEQ + rbase) = pk;
            }
        }
    }
}

// ---------------------------------------------------------------------------
// Kernel 3: flash attention. Block = 8 waves = 256 queries of one (b,h).
// K (32x64) and V^T (64x32) chunks DMA'd into LDS by the Tensor Data Mover
// (one issuing wave, double buffered, s_wait_tensorcnt + barrier), shared by
// all waves. Each wave: 32 queries; per chunk 8 WMMAs scores + online
// softmax + LDS transpose of P + 8 WMMAs P@V.
// ---------------------------------------------------------------------------
__global__ void __launch_bounds__(256, 1)
flash_attn(const _Float16* __restrict__ qh,
           const _Float16* __restrict__ kh,
           const _Float16* __restrict__ vth,
           _Float16* __restrict__ aoh) {
    __shared__ _Float16 kbuf[2][32][72];        // [buf][key][d]  pitch 144B
    __shared__ _Float16 vbuf[2][64][40];        // [buf][d][key]  pitch 80B
    __shared__ _Float16 pbuf[8][32][40];        // per-wave P     pitch 80B

    const int tid   = threadIdx.x;
    const int lane  = tid & 31;
    const int wave  = tid >> 5;
    const int bh    = blockIdx.x >> 3;          // 48 heads
    const int qblk  = blockIdx.x & 7;           // 8 blocks of 256 queries
    const int q0    = qblk * 256 + wave * 32;
    const int lmod  = lane & 15;
    const int khalf = (lane >> 4) * 8;
    _Float16* pl = &pbuf[wave][0][0];

    // persistent Q fragments: 2 query tiles x 2 K-steps (d 0..31, 32..63)
    v16h aq[2][2];
#pragma unroll
    for (int qq = 0; qq < 2; ++qq) {
        const _Float16* qrow =
            qh + ((size_t)(bh * SEQ + q0 + qq * 16 + lmod)) * HDIM;
        aq[qq][0] = load_frag(qrow, khalf);
        aq[qq][1] = load_frag(qrow + 32, khalf);
    }

    const _Float16* gKbase = kh  + (size_t)bh * SEQ * HDIM;   // [key][d]
    const _Float16* gVbase = vth + (size_t)bh * HDIM * SEQ;   // [d][key]

#ifdef HAVE_TDM
    // K: 32 rows x 128B, pad 16B every 128B (interval code 4, amount code 3)
    const i32x8 g1k = { (int)((1u << 16) | (1u << 20) | (4u << 22) | (3u << 25)),
                        (int)(64u << 16),   // tensor_dim0 = 64
                        (int)(32u << 16),   // tensor_dim1 = 32
                        (int)(64u << 16),   // tile_dim0 = 64
                        32,                 // tile_dim1 = 32
                        64, 0, 0 };         // dim0_stride = 64
    // V: 64 rows x 64B, pad 16B every 64B (interval code 3, amount code 3)
    const i32x8 g1v = { (int)((1u << 16) | (1u << 20) | (3u << 22) | (3u << 25)),
                        (int)(32u << 16),   // tensor_dim0 = 32
                        (int)(64u << 16),   // tensor_dim1 = 64
                        (int)(32u << 16),   // tile_dim0 = 32
                        64,                 // tile_dim1 = 64
                        2048, 0, 0 };       // dim0_stride = 2048
    if (wave == 0) {                        // prime buffer 0
        tdm_load_2d((unsigned)(size_t)&kbuf[0][0][0], gKbase, g1k);
        tdm_load_2d((unsigned)(size_t)&vbuf[0][0][0], gVbase, g1v);
    }
#else
    const int kr = tid >> 3, kc = (tid & 7) * 8;
    const int vr = tid >> 2, vc = (tid & 3) * 8;
    const _Float16* gK = gKbase + (size_t)kr * HDIM + kc;
    const _Float16* gV = gVbase + (size_t)vr * SEQ  + vc;
    v8h kreg = *(const v8h*)gK;
    v8h vreg = *(const v8h*)gV;
#endif

    v8f oacc[2][4] = {};
    float mrun[2][8], lrun[2][8];
#pragma unroll
    for (int qq = 0; qq < 2; ++qq)
#pragma unroll
        for (int c = 0; c < 8; ++c) { mrun[qq][c] = -1e30f; lrun[qq][c] = 0.0f; }

    for (int jc = 0; jc < SEQ / 32; ++jc) {
        const int buf = jc & 1;
#ifdef HAVE_TDM
        if (wave == 0) {
#if __has_builtin(__builtin_amdgcn_s_wait_tensorcnt)
            __builtin_amdgcn_s_wait_tensorcnt(0);
#else
            asm volatile("s_wait_tensorcnt 0x0" ::: "memory");
#endif
        }
        __syncthreads();                         // DMA'd chunk visible
        if (jc < SEQ / 32 - 1 && wave == 0) {    // DMA next chunk into buf^1
            tdm_load_2d((unsigned)(size_t)&kbuf[buf ^ 1][0][0],
                        gKbase + (size_t)(jc + 1) * 32 * HDIM, g1k);
            tdm_load_2d((unsigned)(size_t)&vbuf[buf ^ 1][0][0],
                        gVbase + (jc + 1) * 32, g1v);
        }
#else
        *(v8h*)&kbuf[buf][kr][kc] = kreg;
        *(v8h*)&vbuf[buf][vr][vc] = vreg;
        if (jc < SEQ / 32 - 1) {
            kreg = *(const v8h*)(gK + (size_t)(jc + 1) * 32 * HDIM);
            vreg = *(const v8h*)(gV + (jc + 1) * 32);
        }
        __syncthreads();
#endif

        // shared B fragments for scores: 2 key tiles x 2 K-steps
        v16h bk[2][2];
#pragma unroll
        for (int t2 = 0; t2 < 2; ++t2) {
            const _Float16* krow = &kbuf[buf][t2 * 16 + lmod][0];
            bk[t2][0] = load_frag(krow, khalf);
            bk[t2][1] = load_frag(krow + 32, khalf);
        }

#pragma unroll
        for (int qq = 0; qq < 2; ++qq) {
            v8f s0 = {}, s1 = {};
            s0 = wmma_f16(aq[qq][0], bk[0][0], s0);
            s0 = wmma_f16(aq[qq][1], bk[0][1], s0);
            s1 = wmma_f16(aq[qq][0], bk[1][0], s1);
            s1 = wmma_f16(aq[qq][1], bk[1][1], s1);

            // online softmax over this 32-key chunk
#pragma unroll
            for (int c = 0; c < 8; ++c) {
                float v0 = s0[c], v1 = s1[c];
                float mx = fmaxf(v0, v1);
                mx = fmaxf(mx, __shfl_xor(mx, 1, 32));
                mx = fmaxf(mx, __shfl_xor(mx, 2, 32));
                mx = fmaxf(mx, __shfl_xor(mx, 4, 32));
                mx = fmaxf(mx, __shfl_xor(mx, 8, 32));
                const float nm = fmaxf(mrun[qq][c], mx);
                const float sc = __expf(mrun[qq][c] - nm);
                const float p0 = __expf(v0 - nm);
                const float p1 = __expf(v1 - nm);
                float rs = p0 + p1;
                rs += __shfl_xor(rs, 1, 32);
                rs += __shfl_xor(rs, 2, 32);
                rs += __shfl_xor(rs, 4, 32);
                rs += __shfl_xor(rs, 8, 32);
                lrun[qq][c] = lrun[qq][c] * sc + rs;
                mrun[qq][c] = nm;
#pragma unroll
                for (int t = 0; t < 4; ++t) oacc[qq][t][c] *= sc;
                const int r = qq * 16 + c + ((lane >> 4) * 8);
                pl[r * 40 + lmod]      = (_Float16)p0;
                pl[r * 40 + 16 + lmod] = (_Float16)p1;
            }
        }
        asm volatile("s_wait_dscnt 0" ::: "memory");   // cross-lane P RAW

        v16h pa0 = load_frag(pl + (0 * 16 + lmod) * 40, khalf);
        v16h pa1 = load_frag(pl + (1 * 16 + lmod) * 40, khalf);
#pragma unroll
        for (int t = 0; t < 4; ++t) {
            v16h bv = load_frag(&vbuf[buf][t * 16 + lmod][0], khalf);
            oacc[0][t] = wmma_f16(pa0, bv, oacc[0][t]);
            oacc[1][t] = wmma_f16(pa1, bv, oacc[1][t]);
        }
    }

    // normalize, write attention output f16 [b][n][hidden]
    const int b = bh / NHEADS, h = bh % NHEADS;
#pragma unroll
    for (int qq = 0; qq < 2; ++qq)
#pragma unroll
        for (int c = 0; c < 8; ++c) {
            const int n = q0 + qq * 16 + c + ((lane >> 4) * 8);
            const float inv = 1.0f / lrun[qq][c];
#pragma unroll
            for (int t = 0; t < 4; ++t) {
                const int hid = h * HDIM + t * 16 + lmod;
                aoh[((size_t)(b * SEQ + n)) * HIDDEN + hid] =
                    (_Float16)(oacc[qq][t][c] * inv);
            }
        }
}

// ---------------------------------------------------------------------------
// Kernel 4: output projection. M=8192, N=768, K=768.  Same LDS-staged scheme.
// ---------------------------------------------------------------------------
__global__ void __launch_bounds__(256, 1)
proj_gemm(const _Float16* __restrict__ aoh,
          const _Float16* __restrict__ wh,
          const float*    __restrict__ bias,
          float* __restrict__ out) {
    __shared__ _Float16 bsh[2][64][40];
    const int tid   = threadIdx.x;
    const int lane  = tid & 31;
    const int wave  = tid >> 5;
    const int brow  = blockIdx.x % (MTOT / 256);        // 32
    const int bcol  = blockIdx.x / (MTOT / 256);        // 12
    const int m0    = brow * 256 + wave * 32;
    const int o0    = bcol * 64;
    const int lmod  = lane & 15;
    const int khalf = (lane >> 4) * 8;

    const int scol = tid >> 2;
    const int skc  = (tid & 3) * 8;
    const _Float16* gB = wh + (size_t)(o0 + scol) * HIDDEN + skc;
    v8h breg = *(const v8h*)gB;

    const _Float16* ar0 = aoh + (size_t)(m0 + lmod) * HIDDEN;
    const _Float16* ar1 = ar0 + (size_t)16 * HIDDEN;

    v8f acc[2][4] = {};
    for (int kt = 0; kt < 24; ++kt) {
        const int buf = kt & 1;
        *(v8h*)&bsh[buf][scol][skc] = breg;
        if (kt < 23) breg = *(const v8h*)(gB + (kt + 1) * 32);
        __syncthreads();
        const int k = kt * 32;
        v16h a0 = load_frag(ar0 + k, khalf);
        v16h a1 = load_frag(ar1 + k, khalf);
#pragma unroll
        for (int t = 0; t < 4; ++t) {
            v16h b = load_frag(&bsh[buf][t * 16 + lmod][0], khalf);
            acc[0][t] = wmma_f16(a0, b, acc[0][t]);
            acc[1][t] = wmma_f16(a1, b, acc[1][t]);
        }
    }

#pragma unroll
    for (int mt = 0; mt < 2; ++mt) {
        const int rbase = m0 + mt * 16 + ((lane >> 4) * 8);
#pragma unroll
        for (int t = 0; t < 4; ++t) {
            const int o = o0 + t * 16 + lmod;
            const float bv = bias[o];
#pragma unroll
            for (int c = 0; c < 8; ++c)
                out[((size_t)(rbase + c)) * HIDDEN + o] = acc[mt][t][c] + bv;
        }
    }
}

// ---------------------------------------------------------------------------
extern "C" void kernel_launch(void* const* d_in, const int* in_sizes, int n_in,
                              void* d_out, int out_size, void* d_ws, size_t ws_size,
                              hipStream_t stream) {
    const float* x      = (const float*)d_in[0];
    const float* qkv_w  = (const float*)d_in[1];
    const float* qkv_b  = (const float*)d_in[2];
    const float* proj_w = (const float*)d_in[3];
    const float* proj_b = (const float*)d_in[4];
    float* out = (float*)d_out;

    char* ws = (char*)d_ws;
    size_t off = 0;
    _Float16* xh  = (_Float16*)(ws + off); off += (size_t)MTOT * HIDDEN * 2;
    _Float16* wqh = (_Float16*)(ws + off); off += (size_t)QKV_OUT * HIDDEN * 2;
    _Float16* wph = (_Float16*)(ws + off); off += (size_t)HIDDEN * HIDDEN * 2;
    _Float16* qh  = (_Float16*)(ws + off); off += (size_t)BATCH * NHEADS * SEQ * HDIM * 2;
    _Float16* kh  = (_Float16*)(ws + off); off += (size_t)BATCH * NHEADS * SEQ * HDIM * 2;
    _Float16* vth = (_Float16*)(ws + off); off += (size_t)BATCH * NHEADS * SEQ * HDIM * 2;
    _Float16* aoh = (_Float16*)(ws + off); off += (size_t)MTOT * HIDDEN * 2;

    cast_f32_to_f16<<<(MTOT * HIDDEN) / 1024, 256, 0, stream>>>(x, xh, MTOT * HIDDEN);
    cast_f32_to_f16<<<(QKV_OUT * HIDDEN) / 1024, 256, 0, stream>>>(qkv_w, wqh, QKV_OUT * HIDDEN);
    cast_f32_to_f16<<<(HIDDEN * HIDDEN) / 1024, 256, 0, stream>>>(proj_w, wph, HIDDEN * HIDDEN);

    // 32 row-blocks x 36 col-blocks
    qkv_gemm<<<32 * 36, 256, 0, stream>>>(xh, wqh, qkv_b, qh, kh, vth);

    // 48 heads x 8 query-blocks of 256
    flash_attn<<<48 * 8, 256, 0, stream>>>(qh, kh, vth, aoh);

    // 32 row-blocks x 12 col-blocks
    proj_gemm<<<32 * 12, 256, 0, stream>>>(aoh, wph, proj_b, out);
}
